// noiseEdgeConv_19086834664034
// MI455X (gfx1250) — compile-verified, matrix operand order checked by reading, not run on hardware
//
#include <hip/hip_runtime.h>

// ---------------------------------------------------------------------------
// EdgeConv (DGCNN-style) for MI455X / gfx1250, wave32 + WMMA bf16 + async LDS.
//
// Pipeline (all on `stream`, serialized by stream order):
//   1) knn_kernel    : brute-force 32-NN over 8192 pts / batch (3-D coords)
//   2) point_kernel  : g = relu(relu(f W3^T + b3) W4^T + b4)        [WMMA]
//   3) edge_kernel   : x = max_k relu(relu(e W1^T + b1) W2^T + b2)  [WMMA]
//   4) final_kernel  : out = (x + g) W5^T + b5                      [WMMA]
//
// Workspace: [0,4MiB) knn idx int32, [4,20MiB) g fp32, [20,36MiB) x fp32.
// ---------------------------------------------------------------------------

#define BATCH 4
#define NPTS  8192
#define KNNK  32
#define HID   128
#define ODIM  64
#define LDW   136   // LDS row stride in bf16 elems: 272B = 68 dwords (bank-spread, 16B aligned)

typedef __bf16 bf16_t;
typedef __attribute__((ext_vector_type(16))) __bf16 v16bf;
typedef __attribute__((ext_vector_type(8)))  __bf16 v8bf;
typedef __attribute__((ext_vector_type(8)))  float  v8f;

// -------- async global->LDS copy path (ASYNCcnt), guarded for portability ---
#if defined(__gfx1250__) && __has_builtin(__builtin_amdgcn_global_load_async_to_lds_b128) && \
    __has_builtin(__builtin_amdgcn_s_wait_asynccnt)
#define HAVE_ASYNC_LDS 1
typedef __attribute__((ext_vector_type(4))) int i32x4;
typedef __attribute__((address_space(1))) i32x4 gl_i32x4;    // prints as "int4 __device__*"
typedef __attribute__((address_space(3))) i32x4 lds_i32x4;   // prints as "int4 __shared__*"
#else
#define HAVE_ASYNC_LDS 0
#endif

// Copy nfloats (multiple of 4, both pointers 16B aligned) from global to LDS.
// Async path issues ASYNCcnt-tracked global_load_async_to_lds_b128.
__device__ __forceinline__ void stage_f32(float* dst_lds, const float* src,
                                          int nfloats, int tid, int nthreads) {
#if HAVE_ASYNC_LDS
    for (int i = tid * 4; i < nfloats; i += nthreads * 4)
        __builtin_amdgcn_global_load_async_to_lds_b128(
            (gl_i32x4*)(src + i), (lds_i32x4*)(dst_lds + i), 0, 0);
#else
    for (int i = tid; i < nfloats; i += nthreads) dst_lds[i] = src[i];
#endif
}

__device__ __forceinline__ void stage_wait() {
#if HAVE_ASYNC_LDS
    __builtin_amdgcn_s_wait_asynccnt(0);   // this wave's async LDS writes done
#endif
}

// ISA 16-bit matrix fragment K mapping (CDNA5 ISA):
//   lanes 0-15 : elems 0..7 -> K 0..7,   elems 8..15 -> K 16..23
//   lanes 16-31: elems 0..7 -> K 8..15,  elems 8..15 -> K 24..31
__device__ __forceinline__ int frag_k(int e, int hf) {
    int base = hf * 8;
    if (e >= 8) base += 16;
    return base + (e & 7);
}

// Load a 16-wide bf16 fragment row from LDS (two 16B ds_load_b128 chunks).
// 'row' = per-lane M (for A) or N (for B) row, 'k0' = K-step offset (multiple of 32).
__device__ __forceinline__ v16bf lds_frag(const bf16_t* base, int ld, int row, int k0, int hf) {
    const bf16_t* p = base + row * ld + k0 + (hf ? 8 : 0);
    v8bf lo = *(const v8bf*)(p);
    v8bf hi = *(const v8bf*)(p + 16);
    return __builtin_shufflevector(lo, hi, 0, 1, 2, 3, 4, 5, 6, 7,
                                           8, 9, 10, 11, 12, 13, 14, 15);
}

__device__ __forceinline__ v8f wmma_bf16(v16bf a, v16bf b, v8f c) {
    return __builtin_amdgcn_wmma_f32_16x16x32_bf16(false, a, false, b,
                                                   (short)0, c, false, false);
}

// ---------------------------------------------------------------------------
// 1) Brute-force KNN: one query point per thread, candidates tiled in LDS.
//    Candidate tiles are staged with async global->LDS b128 copies.
// ---------------------------------------------------------------------------
__global__ __launch_bounds__(256) void knn_kernel(const float* __restrict__ f,
                                                  int* __restrict__ knn) {
    __shared__ alignas(16) float tile[512 * 3];
    const int b = blockIdx.x >> 5;                       // 32 blocks per batch
    const int q = ((blockIdx.x & 31) << 8) + threadIdx.x;
    const float* fb = f + (size_t)b * NPTS * 3;

    const float qx = fb[q * 3 + 0], qy = fb[q * 3 + 1], qz = fb[q * 3 + 2];
    const float qsq = qx * qx + qy * qy + qz * qz;

    float bestd[KNNK];
    int   besti[KNNK];
    #pragma unroll
    for (int i = 0; i < KNNK; ++i) { bestd[i] = 3.0e38f; besti[i] = 0; }
    float worst = 3.0e38f;
    int   wslot = 0;

    for (int base = 0; base < NPTS; base += 512) {
        __syncthreads();                                  // previous tile fully consumed
        stage_f32(tile, fb + (size_t)base * 3, 512 * 3, threadIdx.x, 256);
        if (base + 512 < NPTS)   // speculative prefetch of next tile -> global_prefetch_b8
            __builtin_prefetch(fb + (size_t)(base + 512) * 3 + threadIdx.x * 6, 0, 0);
        stage_wait();
        __syncthreads();
        for (int c = 0; c < 512; ++c) {
            const float cx = tile[c * 3 + 0], cy = tile[c * 3 + 1], cz = tile[c * 3 + 2];
            const float csq = cx * cx + cy * cy + cz * cz;
            const float d = qsq + csq - 2.0f * (qx * cx + qy * cy + qz * cz);
            if (d < worst) {
                bestd[wslot] = d;
                besti[wslot] = base + c;
                float w = bestd[0]; int s = 0;
                for (int i = 1; i < KNNK; ++i)
                    if (bestd[i] > w) { w = bestd[i]; s = i; }
                worst = w; wslot = s;
            }
        }
    }
    for (int i = 0; i < KNNK; ++i)
        knn[((size_t)b * NPTS + q) * KNNK + i] = besti[i];
}

// ---------------------------------------------------------------------------
// 2) Point branch: g = relu(relu(f W3^T + b3) W4^T + b4).  8 waves x 16 pts.
// ---------------------------------------------------------------------------
__global__ __launch_bounds__(256) void point_kernel(const float* __restrict__ f,
                                                    const float* __restrict__ W3,
                                                    const float* __restrict__ b3,
                                                    const float* __restrict__ W4,
                                                    const float* __restrict__ b4,
                                                    float* __restrict__ g) {
    __shared__ alignas(16) bf16_t w4s[HID * LDW];
    __shared__ alignas(16) float w3s[HID * 3];
    __shared__ alignas(16) float b3s[HID];
    __shared__ alignas(16) float b4s[HID];

    const int tid = threadIdx.x;
    stage_f32(w3s, W3, HID * 3, tid, 256);   // async: pure fp32 copies
    stage_f32(b3s, b3, HID, tid, 256);
    stage_f32(b4s, b4, HID, tid, 256);
    for (int i = tid; i < HID * HID; i += 256) {   // bf16 convert stays on VALU
        const int r = i >> 7, c = i & 127;
        w4s[r * LDW + c] = (bf16_t)W4[i];
    }
    stage_wait();
    __syncthreads();

    const int wave = tid >> 5, lane = tid & 31;
    const int m = lane & 15, hf = lane >> 4;
    const size_t pbase = (size_t)blockIdx.x * 128 + wave * 16;
    const size_t gp = pbase + m;

    const float f0 = f[gp * 3 + 0], f1 = f[gp * 3 + 1], f2 = f[gp * 3 + 2];

    // Layer 3 (3 -> 128) computed by VALU directly into WMMA A fragments.
    v16bf a[4];
    #pragma unroll
    for (int s = 0; s < 4; ++s) {
        #pragma unroll
        for (int e = 0; e < 16; ++e) {
            const int k = s * 32 + frag_k(e, hf);
            float h = b3s[k] + f0 * w3s[k * 3 + 0] + f1 * w3s[k * 3 + 1] + f2 * w3s[k * 3 + 2];
            a[s][e] = (bf16_t)fmaxf(h, 0.0f);
        }
    }

    v8f zero = {0.f, 0.f, 0.f, 0.f, 0.f, 0.f, 0.f, 0.f};
    v8f acc[8];
    #pragma unroll
    for (int t = 0; t < 8; ++t) acc[t] = zero;

    #pragma unroll
    for (int s = 0; s < 4; ++s) {
        #pragma unroll
        for (int t = 0; t < 8; ++t) {
            const v16bf bfr = lds_frag(w4s, LDW, t * 16 + m, s * 32, hf);
            acc[t] = wmma_bf16(a[s], bfr, acc[t]);
        }
    }

    #pragma unroll
    for (int t = 0; t < 8; ++t) {
        const float bias = b4s[t * 16 + m];
        #pragma unroll
        for (int v = 0; v < 8; ++v) {
            const int row = v + hf * 8;   // D layout: lanes 0-15 rows 0..7, lanes 16-31 rows 8..15
            g[(pbase + row) * HID + t * 16 + m] = fmaxf(acc[t][v] + bias, 0.0f);
        }
    }
}

// ---------------------------------------------------------------------------
// 3) Edge branch: one wave per point, 2 M-tiles of 16 neighbors each.
//    x[p][:] = max_k relu(relu(e W1^T + b1) W2^T + b2)
// ---------------------------------------------------------------------------
__global__ __launch_bounds__(256) void edge_kernel(const float* __restrict__ f,
                                                   const int* __restrict__ knn,
                                                   const float* __restrict__ W1,
                                                   const float* __restrict__ b1,
                                                   const float* __restrict__ W2,
                                                   const float* __restrict__ b2,
                                                   float* __restrict__ x) {
    __shared__ alignas(16) bf16_t w2s[HID * LDW];
    __shared__ alignas(16) float w1s[HID * 6];
    __shared__ alignas(16) float b1s[HID];
    __shared__ alignas(16) float b2s[HID];

    const int tid = threadIdx.x;
    stage_f32(w1s, W1, HID * 6, tid, 256);   // async: pure fp32 copies
    stage_f32(b1s, b1, HID, tid, 256);
    stage_f32(b2s, b2, HID, tid, 256);
    for (int i = tid; i < HID * HID; i += 256) {   // bf16 convert stays on VALU
        const int r = i >> 7, c = i & 127;
        w2s[r * LDW + c] = (bf16_t)W2[i];
    }
    stage_wait();
    __syncthreads();

    const int wave = tid >> 5, lane = tid & 31;
    const int m = lane & 15, hf = lane >> 4;
    const size_t P = (size_t)blockIdx.x * 8 + wave;   // flat point index
    const int bb = (int)(P >> 13);                    // batch (N = 8192)
    const int pn = (int)(P & (NPTS - 1));
    const float* fb = f + (size_t)bb * NPTS * 3;

    const float cx = fb[pn * 3 + 0], cy = fb[pn * 3 + 1], cz = fb[pn * 3 + 2];

    float xm[8];
    #pragma unroll
    for (int t = 0; t < 8; ++t) xm[t] = 0.0f;   // relu output >= 0, so 0 is the identity

    v8f zero = {0.f, 0.f, 0.f, 0.f, 0.f, 0.f, 0.f, 0.f};

    for (int mt = 0; mt < 2; ++mt) {
        // This lane's edge row: neighbor slot j of point P.
        const int j  = mt * 16 + m;
        const int nb = knn[P * KNNK + j];
        const float nx = fb[nb * 3 + 0], ny = fb[nb * 3 + 1], nz = fb[nb * 3 + 2];
        float e6[6] = {nx, ny, nz, nx - cx, ny - cy, nz - cz};

        // Layer 1 (6 -> 128) via VALU, written directly into A fragments
        // (co-executes with XDL WMMA; only ~4.5% of total FLOPs).
        v16bf a[4];
        #pragma unroll
        for (int s = 0; s < 4; ++s) {
            #pragma unroll
            for (int e = 0; e < 16; ++e) {
                const int k = s * 32 + frag_k(e, hf);
                float h = b1s[k];
                #pragma unroll
                for (int c = 0; c < 6; ++c) h += e6[c] * w1s[k * 6 + c];
                a[s][e] = (bf16_t)fmaxf(h, 0.0f);
            }
        }

        // Layer 2 (128 -> 128): 4 K-steps x 8 N-tiles of WMMA bf16.
        v8f acc[8];
        #pragma unroll
        for (int t = 0; t < 8; ++t) acc[t] = zero;
        #pragma unroll
        for (int s = 0; s < 4; ++s) {
            #pragma unroll
            for (int t = 0; t < 8; ++t) {
                const v16bf bfr = lds_frag(w2s, LDW, t * 16 + m, s * 32, hf);
                acc[t] = wmma_bf16(a[s], bfr, acc[t]);
            }
        }

        // Fused b2 + relu + max-pool over this tile's 16 rows.
        #pragma unroll
        for (int t = 0; t < 8; ++t) {
            const float bias = b2s[t * 16 + m];
            float cm = 0.0f;                       // starts at 0 == relu floor
            #pragma unroll
            for (int v = 0; v < 8; ++v) cm = fmaxf(cm, acc[t][v] + bias);
            cm = fmaxf(cm, __shfl_xor(cm, 16));    // combine the two 8-row halves (wave32)
            xm[t] = fmaxf(xm[t], cm);
        }
    }

    if (hf == 0) {   // lanes 0..15 hold column maxes for cols 16t+m
        #pragma unroll
        for (int t = 0; t < 8; ++t)
            x[P * HID + t * 16 + m] = xm[t];
    }
}

// ---------------------------------------------------------------------------
// 4) Final projection: out = (x + g) W5^T + b5.   8 waves x 16 points.
// ---------------------------------------------------------------------------
__global__ __launch_bounds__(256) void final_kernel(const float* __restrict__ xbuf,
                                                    const float* __restrict__ gbuf,
                                                    const float* __restrict__ W5,
                                                    const float* __restrict__ b5,
                                                    float* __restrict__ out) {
    __shared__ alignas(16) bf16_t as_[128 * LDW];   // (x+g) for 128 points, bf16
    __shared__ alignas(16) bf16_t w5s[ODIM * LDW];
    __shared__ alignas(16) float b5s[ODIM];

    const int tid = threadIdx.x;
    const size_t pbase = (size_t)blockIdx.x * 128;

    stage_f32(b5s, b5, ODIM, tid, 256);      // async fp32 copy
    for (int i = tid; i < 128 * HID; i += 256) {
        const int r = i >> 7, c = i & 127;
        const size_t o = (pbase + r) * HID + c;
        as_[r * LDW + c] = (bf16_t)(xbuf[o] + gbuf[o]);
    }
    for (int i = tid; i < ODIM * HID; i += 256) {
        const int r = i >> 7, c = i & 127;
        w5s[r * LDW + c] = (bf16_t)W5[i];
    }
    stage_wait();
    __syncthreads();

    const int wave = tid >> 5, lane = tid & 31;
    const int m = lane & 15, hf = lane >> 4;

    v8f zero = {0.f, 0.f, 0.f, 0.f, 0.f, 0.f, 0.f, 0.f};
    v8f acc[4];
    #pragma unroll
    for (int t = 0; t < 4; ++t) acc[t] = zero;

    #pragma unroll
    for (int s = 0; s < 4; ++s) {
        const v16bf av = lds_frag(as_, LDW, wave * 16 + m, s * 32, hf);
        #pragma unroll
        for (int t = 0; t < 4; ++t) {
            const v16bf bv = lds_frag(w5s, LDW, t * 16 + m, s * 32, hf);
            acc[t] = wmma_bf16(av, bv, acc[t]);
        }
    }

    #pragma unroll
    for (int t = 0; t < 4; ++t) {
        const float bias = b5s[t * 16 + m];
        #pragma unroll
        for (int v = 0; v < 8; ++v) {
            const int row = v + hf * 8;
            out[(pbase + wave * 16 + row) * ODIM + t * 16 + m] = acc[t][v] + bias;
        }
    }
}

// ---------------------------------------------------------------------------
extern "C" void kernel_launch(void* const* d_in, const int* in_sizes, int n_in,
                              void* d_out, int out_size, void* d_ws, size_t ws_size,
                              hipStream_t stream) {
    (void)in_sizes; (void)n_in; (void)out_size; (void)ws_size;

    const float* f  = (const float*)d_in[0];
    const float* W1 = (const float*)d_in[1];
    const float* b1 = (const float*)d_in[2];
    const float* W2 = (const float*)d_in[3];
    const float* b2 = (const float*)d_in[4];
    const float* W3 = (const float*)d_in[5];
    const float* b3 = (const float*)d_in[6];
    const float* W4 = (const float*)d_in[7];
    const float* b4 = (const float*)d_in[8];
    const float* W5 = (const float*)d_in[9];
    const float* b5 = (const float*)d_in[10];
    float* out = (float*)d_out;

    char* ws = (char*)d_ws;
    int*   knn  = (int*)ws;                                   //  4 MiB
    float* gbuf = (float*)(ws + (size_t)4  * 1024 * 1024);    // 16 MiB
    float* xbuf = (float*)(ws + (size_t)20 * 1024 * 1024);    // 16 MiB

    knn_kernel  <<<BATCH * (NPTS / 256), 256, 0, stream>>>(f, knn);
    point_kernel<<<BATCH * NPTS / 128,   256, 0, stream>>>(f, W3, b3, W4, b4, gbuf);
    edge_kernel <<<BATCH * NPTS / 8,     256, 0, stream>>>(f, knn, W1, b1, W2, b2, xbuf);
    final_kernel<<<BATCH * NPTS / 128,   256, 0, stream>>>(xbuf, gbuf, W5, b5, out);
}
